// ImagePoolingAttn_1494648619582
// MI455X (gfx1250) — matrix operand order repaired and checked by reading, not compiled, not run on hardware
//
#include <hip/hip_runtime.h>
#include <math.h>

// ---------------------------------------------------------------------------
// ImagePoolingAttn for MI455X (gfx1250, wave32).
// Big GEMMs (1x1 convs, Q-proj, out-proj) use v_wmma_f32_16x16x32_f16 with
// f32 accumulation. Weights are pre-converted/transposed to f16 K-contiguous
// layouts once; hot-loop staging is either packed f32->f16 converts (data
// that must be converted) or GLOBAL_LOAD_ASYNC_TO_LDS copies (pure f16
// copies, tracked by ASYNCcnt). Roofline: ~230 MB @ 23.3 TB/s (~10us)
// dominates 26 GFLOP of f16 matrix math -> design for the HBM bound.
// ---------------------------------------------------------------------------

#define BS   32
#define EC   256
#define CT   512
#define NH   8
#define HC   32
#define NTOK 27
#define NTXT 512

typedef _Float16 h16;
typedef __attribute__((ext_vector_type(16))) _Float16 v16h;
typedef __attribute__((ext_vector_type(8)))  _Float16 v8h;
typedef __attribute__((ext_vector_type(4)))  _Float16 v4h;
typedef __attribute__((ext_vector_type(8)))  float    v8f;

union pk2 { h16 h[2]; unsigned u; };

__device__ __forceinline__ v16h cat8(v8h a, v8h b) {
  return __builtin_shufflevector(a, b, 0,1,2,3,4,5,6,7,8,9,10,11,12,13,14,15);
}
// A-matrix 16x32 f16 fragment (ISA 7.12.2): lane L<16 -> M=L, half=L>>4.
// elems 0..7  = K[half*8 .. half*8+7], elems 8..15 = K[16+half*8 ..].
__device__ __forceinline__ v16h load_a_frag(const h16* row, int half) {
  v8h lo = *(const v8h*)(row + half * 8);
  v8h hi = *(const v8h*)(row + 16 + half * 8);
  return cat8(lo, hi);
}
// B-matrix 32x16: staged N-major (bt[n][k]); lane n=L&15, elems 0..7 -> K=half*8+e, elems 8..15 -> K=16+half*8+(e-8).
__device__ __forceinline__ v16h load_b_frag(const h16* row, int half) {
  v8h lo = *(const v8h*)(row + half * 8);
  v8h hi = *(const v8h*)(row + 16 + half * 8);
  return cat8(lo, hi);
}
__device__ __forceinline__ v8f wmma16(v16h a, v16h b, v8f c) {
  return __builtin_amdgcn_wmma_f32_16x16x32_f16(false, a, false, b,
                                                (short)0, c, false, false);
}

// --- CDNA5 async global->LDS copies (ASYNCcnt-tracked, no VGPR round-trip) --
__device__ __forceinline__ unsigned lds_off32(const void* p) {
  // Flat shared-aperture address: LDS offset = addr[31:0] (ISA 10.2).
  return (unsigned)(unsigned long long)p;
}
__device__ __forceinline__ void async_b128(unsigned ldsoff, const void* g) {
  asm volatile("global_load_async_to_lds_b128 %0, %1, off"
               :: "v"(ldsoff), "v"(g) : "memory");
}
__device__ __forceinline__ void async_b64(unsigned ldsoff, const void* g) {
  asm volatile("global_load_async_to_lds_b64 %0, %1, off"
               :: "v"(ldsoff), "v"(g) : "memory");
}
__device__ __forceinline__ void wait_async() {
  asm volatile("s_wait_asynccnt 0" ::: "memory");
}

// ---------------------------------------------------------------------------
// Kernel P: convert+transpose weights f32[K][E] -> f16[E][K] (K contiguous).
// ---------------------------------------------------------------------------
__global__ void k_cvt_t(const float* __restrict__ in, h16* __restrict__ outp,
                        int E, int K) {
  int i = blockIdx.x * blockDim.x + threadIdx.x;
  if (i < E * K) {
    int e = i / K, k = i - e * K;
    outp[i] = (h16)in[(size_t)k * E + e];
  }
}

// ---------------------------------------------------------------------------
// Kernel 0: init pooled-feature buffer to -inf (combined via atomic fmax).
// ---------------------------------------------------------------------------
__global__ void k_init_xc(float* __restrict__ xc, int n) {
  int i = blockIdx.x * blockDim.x + threadIdx.x;
  if (i < n) xc[i] = -1.0e30f;
}

// ---------------------------------------------------------------------------
// Kernel 1: fused 1x1-conv (GEMM) + adaptive max pool (3x3 bins).
// Grid: BS * 9 bins * chunks.  Block: 512 threads (16 waves).
// A = Wt (f16, pre-transposed [e][k]) loaded straight from global per lane
// (zero intra-WG reuse -> LDS staging would be pure overhead).
// B = x pixels staged in LDS as f16 (paired k converts, b32 stores).
// Epilogue: max over pixel tiles in-register, shfl-max over the 16 n-lanes,
// then global atomic fmax into xc[b][t][e].
// ---------------------------------------------------------------------------
template <int NT, int BSIDE>
__global__ void __launch_bounds__(512)
k_conv_pool(const float* __restrict__ x, const h16* __restrict__ wt,
            float* __restrict__ xc, int C, int H, int W, int chunks, int tbase)
{
  __shared__ h16 lds_bt[NT * 16][32];    // <= 8 KB
  constexpr int NPX = NT * 16;

  const int tid = threadIdx.x;
  int id = blockIdx.x;
  const int b = id / (9 * chunks); id -= b * 9 * chunks;
  const int bin = id / chunks;
  const int chunk = id - bin * chunks;
  const int r0 = (bin / 3) * BSIDE, c0 = (bin % 3) * BSIDE;
  const int wave = tid >> 5, lane = tid & 31, half = lane >> 4, mn = lane & 15;

  const h16* arow = wt + (size_t)(wave * 16 + mn) * C;   // this lane's A row

  const v8f vz = {0.f,0.f,0.f,0.f,0.f,0.f,0.f,0.f};
  v8f acc[NT];
#pragma unroll
  for (int nt = 0; nt < NT; ++nt) acc[nt] = vz;

  for (int k0 = 0; k0 < C; k0 += 32) {
    __syncthreads();
    // Stage B: pairs along k -> one packed b32 LDS store.
    for (int idx = tid; idx < NPX * 16; idx += 512) {
      int n = idx % NPX, kp = idx / NPX;
      int kk = kp * 2;
      int p = chunk * NPX + n;
      int pr = p / BSIDE, pc = p % BSIDE;
      size_t base = (((size_t)b * C + (k0 + kk)) * H + (r0 + pr)) * W + (c0 + pc);
      pk2 pk;
      pk.h[0] = (h16)x[base];
      pk.h[1] = (h16)x[base + (size_t)H * W];
      *(unsigned*)&lds_bt[n][kk] = pk.u;
    }
    __syncthreads();
    v16h af = cat8(*(const v8h*)(arow + k0 + half * 8),
                   *(const v8h*)(arow + k0 + 16 + half * 8));
#pragma unroll
    for (int nt = 0; nt < NT; ++nt) {
      v16h bf = load_b_frag(&lds_bt[nt * 16 + mn][0], half);
      acc[nt] = wmma16(af, bf, acc[nt]);
    }
  }

  // Max over the wave's pixel tiles (N dim), then over the 16 n-lanes.
  float rmax[8];
#pragma unroll
  for (int r = 0; r < 8; ++r) {
    float m = acc[0][r];
#pragma unroll
    for (int nt = 1; nt < NT; ++nt) m = fmaxf(m, acc[nt][r]);
    rmax[r] = m;
  }
#pragma unroll
  for (int off = 8; off >= 1; off >>= 1) {
#pragma unroll
    for (int r = 0; r < 8; ++r)
      rmax[r] = fmaxf(rmax[r], __shfl_xor(rmax[r], off, 16));
  }
  if (mn == 0) {
    const int t = tbase + bin;
    float* dst = xc + ((size_t)b * NTOK + t) * EC + wave * 16 + half * 8;
#pragma unroll
    for (int r = 0; r < 8; ++r)
      __hip_atomic_fetch_max(dst + r, rmax[r], __ATOMIC_RELAXED,
                             __HIP_MEMORY_SCOPE_AGENT);
  }
}

// ---------------------------------------------------------------------------
// Kernel 2: K or V projection.  xc(+conv bias) -> LN -> @ W + b.
// Tiny (27x256 @ 256x256 per batch): per-wave row LN, then loop-interchanged
// VALU GEMM (27 register accumulators, wm loaded once per c).
// ---------------------------------------------------------------------------
__global__ void __launch_bounds__(256)
k_kv(const float* __restrict__ xc,
     const float* __restrict__ cb0, const float* __restrict__ cb1,
     const float* __restrict__ cb2,
     const float* __restrict__ lng, const float* __restrict__ lnb,
     const float* __restrict__ wm, const float* __restrict__ bias,
     float* __restrict__ out)
{
  __shared__ float lds_ln[NTOK][EC];
  const int b = blockIdx.x;
  const int tid = threadIdx.x, wave = tid >> 5, lane = tid & 31;

  for (int j = 0; j < 4; ++j) {
    int t = wave + 8 * j;
    if (t < NTOK) {
      const float* cbp = (t < 9) ? cb0 : ((t < 18) ? cb1 : cb2);
      float rv[8]; float s = 0.f, sq = 0.f;
#pragma unroll
      for (int i = 0; i < 8; ++i) {
        int c = lane + 32 * i;
        float v = xc[((size_t)b * NTOK + t) * EC + c] + cbp[c];
        rv[i] = v; s += v; sq += v * v;
      }
#pragma unroll
      for (int off = 16; off >= 1; off >>= 1) {
        s  += __shfl_xor(s,  off, 32);
        sq += __shfl_xor(sq, off, 32);
      }
      float mean = s * (1.f / EC);
      float var  = sq * (1.f / EC) - mean * mean;
      float rstd = rsqrtf(var + 1e-5f);
#pragma unroll
      for (int i = 0; i < 8; ++i) {
        int c = lane + 32 * i;
        lds_ln[t][c] = (rv[i] - mean) * rstd * lng[c] + lnb[c];
      }
    }
  }
  __syncthreads();
  const int e = tid;
  float accv[NTOK];
#pragma unroll
  for (int t = 0; t < NTOK; ++t) accv[t] = bias[e];
  for (int c = 0; c < EC; ++c) {
    float wv = wm[(size_t)c * EC + e];
#pragma unroll
    for (int t = 0; t < NTOK; ++t) accv[t] += lds_ln[t][c] * wv;
  }
#pragma unroll
  for (int t = 0; t < NTOK; ++t)
    out[((size_t)b * NTOK + t) * EC + e] = accv[t];
}

// ---------------------------------------------------------------------------
// Kernel 3: Q = LN(text) @ q_w + q_b.  Grid (BS, 8): 64-row tile per WG.
// Row stats fused; A staged with paired LN converts (b64 load, b32 ds store);
// B staged with async global->LDS b128 copies from pre-transposed qwt[n][k].
// ---------------------------------------------------------------------------
__global__ void __launch_bounds__(512)
k_qproj(const float* __restrict__ text,
        const float* __restrict__ lng, const float* __restrict__ lnb,
        const h16* __restrict__ qwt, const float* __restrict__ qb,
        float* __restrict__ qout)
{
  __shared__ h16 lds_a[64][32];
  __shared__ h16 lds_bt[EC][32];
  __shared__ float ps[64][8], pq[64][8];
  __shared__ float smean[64], srstd[64];
  const int b = blockIdx.x, rb = blockIdx.y * 64;
  const int tid = threadIdx.x;

  { // per-row mean/var over CT=512, 8 threads per row
    int r = tid >> 3, s8 = tid & 7;
    const float* row = text + ((size_t)b * NTXT + rb + r) * CT;
    float s = 0.f, sq = 0.f;
    for (int j = 0; j < 64; ++j) { float v = row[s8 + 8 * j]; s += v; sq += v * v; }
    ps[r][s8] = s; pq[r][s8] = sq;
  }
  __syncthreads();
  if (tid < 64) {
    float s = 0.f, sq = 0.f;
    for (int j = 0; j < 8; ++j) { s += ps[tid][j]; sq += pq[tid][j]; }
    float mean = s * (1.f / CT);
    float var  = sq * (1.f / CT) - mean * mean;
    smean[tid] = mean; srstd[tid] = rsqrtf(var + 1e-5f);
  }

  const int wave = tid >> 5, lane = tid & 31, half = lane >> 4, mn = lane & 15;
  const int mt = wave & 3, ng = wave >> 2;
  const v8f vz = {0.f,0.f,0.f,0.f,0.f,0.f,0.f,0.f};
  v8f acc[4] = {vz, vz, vz, vz};

  for (int k0 = 0; k0 < CT; k0 += 32) {
    __syncthreads();
#pragma unroll
    for (int i = 0; i < 2; ++i) {           // B: qwt 256x32, async b128 copies
      int idx = tid + i * 512;
      int n = idx >> 2, ck = (idx & 3) * 8;
      async_b128(lds_off32(&lds_bt[n][ck]), &qwt[(size_t)n * CT + k0 + ck]);
    }
#pragma unroll
    for (int i = 0; i < 2; ++i) {           // A: LN(text) 64x32, paired k
      int idx = tid + i * 512;
      int r = idx >> 4, kk = (idx & 15) * 2;
      const float* tp = &text[((size_t)b * NTXT + rb + r) * CT + k0 + kk];
      float m = smean[r], rs = srstd[r];
      pk2 pk;
      pk.h[0] = (h16)((tp[0] - m) * rs * lng[k0 + kk]     + lnb[k0 + kk]);
      pk.h[1] = (h16)((tp[1] - m) * rs * lng[k0 + kk + 1] + lnb[k0 + kk + 1]);
      *(unsigned*)&lds_a[r][kk] = pk.u;
    }
    wait_async();
    __syncthreads();
    v16h af = load_a_frag(&lds_a[mt * 16 + mn][0], half);
#pragma unroll
    for (int nt = 0; nt < 4; ++nt) {
      v16h bf = load_b_frag(&lds_bt[ng * 64 + nt * 16 + mn][0], half);
      acc[nt] = wmma16(af, bf, acc[nt]);
    }
  }
#pragma unroll
  for (int nt = 0; nt < 4; ++nt) {
#pragma unroll
    for (int r = 0; r < 8; ++r) {
      int row = rb + mt * 16 + half * 8 + r;
      int col = ng * 64 + nt * 16 + mn;
      qout[((size_t)b * NTXT + row) * EC + col] = acc[nt][r] + qb[col];
    }
  }
}

// ---------------------------------------------------------------------------
// Kernel 4: attention, 27 keys per (batch, head).  Pure VALU; K/V in LDS.
// Output written as f16 (packed pairs) for the proj GEMM A operand.
// ---------------------------------------------------------------------------
__global__ void __launch_bounds__(256)
k_attn(const float* __restrict__ q, const float* __restrict__ kk,
       const float* __restrict__ vv, h16* __restrict__ o)
{
  __shared__ float kh[NTOK][HC], vh[NTOK][HC];
  const int b = blockIdx.x, h = blockIdx.y;
  const int tid = threadIdx.x;
  for (int i = tid; i < NTOK * HC; i += 256) {
    int t = i >> 5, c = i & 31;
    kh[t][c] = kk[((size_t)b * NTOK + t) * EC + h * HC + c];
    vh[t][c] = vv[((size_t)b * NTOK + t) * EC + h * HC + c];
  }
  __syncthreads();
#pragma unroll
  for (int rep = 0; rep < 2; ++rep) {
    int n = tid + rep * 256;
    const float* qrow = q + ((size_t)b * NTXT + n) * EC + h * HC;
    float qv[HC];
#pragma unroll
    for (int c = 0; c < HC; ++c) qv[c] = qrow[c];
    float sc[NTOK]; float mx = -1e30f;
#pragma unroll
    for (int t = 0; t < NTOK; ++t) {
      float s = 0.f;
#pragma unroll
      for (int c = 0; c < HC; ++c) s += qv[c] * kh[t][c];
      s *= 0.17677669529663689f;   // 1/sqrt(32)
      sc[t] = s; mx = fmaxf(mx, s);
    }
    float den = 0.f;
#pragma unroll
    for (int t = 0; t < NTOK; ++t) { float e = __expf(sc[t] - mx); sc[t] = e; den += e; }
    float inv = 1.f / den;
    float res[HC];
#pragma unroll
    for (int c = 0; c < HC; ++c) {
      float s = 0.f;
#pragma unroll
      for (int t = 0; t < NTOK; ++t) s += sc[t] * vh[t][c];
      res[c] = s * inv;
    }
    h16* orow = o + ((size_t)b * NTXT + n) * EC + h * HC;
#pragma unroll
    for (int c = 0; c < HC; c += 2) {
      pk2 pk; pk.h[0] = (h16)res[c]; pk.h[1] = (h16)res[c + 1];
      *(unsigned*)&orow[c] = pk.u;
    }
  }
}

// ---------------------------------------------------------------------------
// Kernel 5: out = o @ proj_w + proj_b + text.  WMMA GEMM, residual epilogue.
// A (o, f16) and B (pre-transposed pwt) staged with async global->LDS copies:
// the whole K-loop has no VGPR data round-trips except the WMMA fragments.
// Grid (BS, 8 row-groups, 4 col-groups); wave: M-tile w&3, 2 N-tiles at w>>2.
// ---------------------------------------------------------------------------
__global__ void __launch_bounds__(512)
k_proj(const h16* __restrict__ o, const h16* __restrict__ pwt,
       const float* __restrict__ pb, const float* __restrict__ text,
       float* __restrict__ out)
{
  __shared__ h16 lds_a[64][32];
  __shared__ h16 lds_bt[128][32];
  const int b = blockIdx.x, mg = blockIdx.y, ngb = blockIdx.z;
  const int tid = threadIdx.x;
  const int wave = tid >> 5, lane = tid & 31, half = lane >> 4, mn = lane & 15;
  const int mt = wave & 3, np = wave >> 2;
  const v8f vz = {0.f,0.f,0.f,0.f,0.f,0.f,0.f,0.f};
  v8f acc[2] = {vz, vz};

  for (int k0 = 0; k0 < EC; k0 += 32) {
    __syncthreads();
    { // A: o 64x32 f16, async b64 (2048 halves = 512 x 8B)
      int r = tid >> 3, ck = (tid & 7) * 4;
      async_b64(lds_off32(&lds_a[r][ck]),
                &o[((size_t)b * NTXT + mg * 64 + r) * EC + k0 + ck]);
    }
    { // B: pwt 128x32 f16, async b128 (4096 halves = 512 x 16B)
      int n = tid >> 2, ck = (tid & 3) * 8;
      async_b128(lds_off32(&lds_bt[n][ck]),
                 &pwt[(size_t)(ngb * 128 + n) * EC + k0 + ck]);
    }
    wait_async();
    __syncthreads();
    v16h af = load_a_frag(&lds_a[mt * 16 + mn][0], half);
#pragma unroll
    for (int nt = 0; nt < 2; ++nt) {
      v16h bf = load_b_frag(&lds_bt[np * 32 + nt * 16 + mn][0], half);
      acc[nt] = wmma16(af, bf, acc[nt]);
    }
  }
#pragma unroll
  for (int nt = 0; nt < 2; ++nt) {
#pragma unroll
    for (int r = 0; r < 8; ++r) {
      int row = mg * 64 + mt * 16 + half * 8 + r;
      int col = ngb * 128 + np * 32 + nt * 16 + mn;
      size_t idx = ((size_t)b * NTXT + row) * CT + col;
      out[idx] = acc[nt][r] + pb[col] + text[idx];
    }
  }
}

// ---------------------------------------------------------------------------
extern "C" void kernel_launch(void* const* d_in, const int* in_sizes, int n_in,
                              void* d_out, int out_size, void* d_ws, size_t ws_size,
                              hipStream_t stream)
{
  (void)in_sizes; (void)n_in; (void)out_size; (void)ws_size;
  const float* x0     = (const float*)d_in[0];
  const float* x1     = (const float*)d_in[1];
  const float* x2     = (const float*)d_in[2];
  const float* text   = (const float*)d_in[3];
  const float* w0     = (const float*)d_in[4];
  const float* cb0    = (const float*)d_in[5];
  const float* w1     = (const float*)d_in[6];
  const float* cb1    = (const float*)d_in[7];
  const float* w2     = (const float*)d_in[8];
  const float* cb2    = (const float*)d_in[9];
  const float* q_ln_g = (const float*)d_in[10];
  const float* q_ln_b = (const float*)d_in[11];
  const float* k_ln_g = (const float*)d_in[12];
  const float* k_ln_b = (const float*)d_in[13];
  const float* v_ln_g = (const float*)d_in[14];
  const float* v_ln_b = (const float*)d_in[15];
  const float* q_w    = (const float*)d_in[16];
  const float* q_b    = (const float*)d_in[17];
  const float* k_w    = (const float*)d_in[18];
  const float* k_b    = (const float*)d_in[19];
  const float* v_w    = (const float*)d_in[20];
  const float* v_b    = (const float*)d_in[21];
  const float* proj_w = (const float*)d_in[22];
  const float* proj_b = (const float*)d_in[23];

  float* out = (float*)d_out;
  float* ws  = (float*)d_ws;
  // f32 region
  float* xc  = ws;                     // 221184
  float* kws = xc  + 221184;           // 221184
  float* vws = kws + 221184;           // 221184
  float* qws = vws + 221184;           // 4194304
  // f16 region (starts 16B-aligned: 4857856*4 bytes)
  h16* ows = (h16*)(qws + 4194304);    // 4194304 halves
  h16* wt0 = ows + 4194304;            //  65536  (256x256)
  h16* wt1 = wt0 + 65536;              // 131072  (256x512)
  h16* wt2 = wt1 + 131072;             // 262144  (256x1024)
  h16* qwt = wt2 + 262144;             // 131072  (256x512)
  h16* pwt = qwt + 131072;             // 131072  (512x256)

  // Pre-convert/transpose weights to f16 K-contiguous layouts.
  k_cvt_t<<<(256 * 256  + 255) / 256, 256, 0, stream>>>(w0,     wt0, 256, 256);
  k_cvt_t<<<(256 * 512  + 255) / 256, 256, 0, stream>>>(w1,     wt1, 256, 512);
  k_cvt_t<<<(256 * 1024 + 255) / 256, 256, 0, stream>>>(w2,     wt2, 256, 1024);
  k_cvt_t<<<(256 * 512  + 255) / 256, 256, 0, stream>>>(q_w,    qwt, 256, 512);
  k_cvt_t<<<(512 * 256  + 255) / 256, 256, 0, stream>>>(proj_w, pwt, 512, 256);

  k_init_xc<<<864, 256, 0, stream>>>(xc, BS * NTOK * EC);

  // conv+pool: scale0 two 128-px chunks/bin; scales 1,2 whole bin.
  k_conv_pool<8, 16><<<BS * 9 * 2, 512, 0, stream>>>(x0, wt0, xc, 256,  48, 48, 2, 0);
  k_conv_pool<4,  8><<<BS * 9,     512, 0, stream>>>(x1, wt1, xc, 512,  24, 24, 1, 9);
  k_conv_pool<1,  4><<<BS * 9,     512, 0, stream>>>(x2, wt2, xc, 1024, 12, 12, 1, 18);

  k_kv<<<BS, 256, 0, stream>>>(xc, cb0, cb1, cb2, k_ln_g, k_ln_b, k_w, k_b, kws);
  k_kv<<<BS, 256, 0, stream>>>(xc, cb0, cb1, cb2, v_ln_g, v_ln_b, v_w, v_b, vws);

  k_qproj<<<dim3(BS, 8), 512, 0, stream>>>(text, q_ln_g, q_ln_b, qwt, q_b, qws);
  k_attn<<<dim3(BS, NH), 256, 0, stream>>>(qws, kws, vws, ows);
  k_proj<<<dim3(BS, 8, 4), 512, 0, stream>>>(ows, pwt, proj_b, text, out);
}
